// GlobalAttentionTransformer_67817533604129
// MI455X (gfx1250) — matrix operand order repaired and checked by reading, not compiled, not run on hardware
//
#include <hip/hip_runtime.h>
#include <hip/hip_bf16.h>

typedef __attribute__((ext_vector_type(16))) _Float16 v16h;
typedef __attribute__((ext_vector_type(8)))  float    v8f;

#define BATCH   8
#define HEADS   8
#define G_TOK   5
#define DMODEL  512
#define DFF     2048
#define S_TOT   4101          // G + 64*64
#define SPAD    4112          // 257 * 16
#define MTILES  257
#define KT1     16            // 512 / 32
#define KT2     64            // 2048 / 32
#define CHUNK_KT 16           // 16 kt-tiles = 16KB LDS staging chunk
#define FM_SIZE ((size_t)BATCH * DMODEL * 4096)

// ---- feature detection (device pass only; host pass sees 0 and skips bodies) ----
#if defined(__has_builtin)
#  if __has_builtin(__builtin_amdgcn_tensor_load_to_lds)
#    define HAVE_TDM 1
#  endif
#  if __has_builtin(__builtin_amdgcn_s_wait_tensorcnt)
#    define HAVE_WAIT_TENSOR 1
#  endif
#endif
#ifndef HAVE_TDM
#  define HAVE_TDM 0
#endif
#ifndef HAVE_WAIT_TENSOR
#  define HAVE_WAIT_TENSOR 0
#endif

// ---------- ISA fragment index helpers (CDNA5 WMMA 16-bit layouts) ----------
// A matrix 16x32 f16: lane<16 holds row r, K{0..7,16..23}; lane>=16 row r, K{8..15,24..31}
__device__ __forceinline__ int afrag_idx(int r, int k) {
  int grp  = k >> 3;                       // 0..3
  int lane = r + ((grp & 1) << 4);
  int e    = (k & 7) + ((grp >> 1) << 3);
  return lane * 16 + e;
}
// B matrix 32x16 f16: lane<16 holds col n, K{0..7,16..23}; lane>=16 col n, K{8..15,24..31}
__device__ __forceinline__ int bfrag_idx(int k, int n) {
  int lane = n + (((k >> 3) & 1) << 4);
  int e    = (k & 7) + ((k >> 4) << 3);
  return lane * 16 + e;
}
__device__ __forceinline__ size_t afrag_base(int b, int mt, int kt, int KT) {
  return (((size_t)b * MTILES + mt) * KT + kt) << 9;   // *512 halfs per 16x32 tile
}

__device__ __forceinline__ float block_reduce_sum256(float v, float* red) {
  int tid = threadIdx.x;
  red[tid] = v; __syncthreads();
  for (int off = 128; off > 0; off >>= 1) {
    if (tid < off) red[tid] += red[tid + off];
    __syncthreads();
  }
  float r = red[0]; __syncthreads();
  return r;
}

// ---------------- TDM: 16KB contiguous global -> LDS (1-D descriptor) ----------------
#if HAVE_TDM
typedef __attribute__((ext_vector_type(4))) unsigned int u32x4;
typedef __attribute__((ext_vector_type(8))) int          i32x8;
typedef __attribute__((ext_vector_type(4))) int          i32x4;

__device__ __forceinline__ void tdm_load_a16k(const _Float16* gsrc, _Float16* ldst) {
  unsigned lds = (unsigned)(size_t)(void*)ldst;                   // LDS offset = low 32b
  unsigned long long ga = (unsigned long long)(size_t)(const void*)gsrc;
  u32x4 g0 = { 1u,                                                // count=1 (user desc)
               lds,                                               // lds_addr [63:32]
               (unsigned)ga,                                      // global_addr[31:0]
               ((unsigned)((ga >> 32) & 0x01FFFFFFu)) | (2u << 30) }; // ga[56:32] | type=2
  i32x8 g1 = { (int)(1u << 16),          // data_size=1 (2 bytes/elem)
               (int)(8192u << 16),       // tensor_dim0 lo16 = 8192
               (int)(1u << 16),          // tensor_dim0 hi=0 | tensor_dim1 lo16 = 1
               (int)(8192u << 16),       // tensor_dim1 hi=0 | tile_dim0 = 8192
               0,                        // tile_dim1=0, tile_dim2=0 (unused)
               8192,                     // tensor_dim0_stride lo32
               0, 0 };                   // stride0 hi / stride1 (unused)
  i32x4 gz4 = {0, 0, 0, 0};
#if defined(__clang_major__) && (__clang_major__ >= 23)
  i32x8 gz8 = {0, 0, 0, 0, 0, 0, 0, 0};
  __builtin_amdgcn_tensor_load_to_lds(g0, g1, gz4, gz4, gz8, 0);
#else
  __builtin_amdgcn_tensor_load_to_lds(g0, g1, gz4, gz4, 0);
#endif
}
#else
__device__ __forceinline__ void coop_copy16k(const _Float16* __restrict__ g,
                                             _Float16* __restrict__ l) {
  const float4* gs = (const float4*)g;
  float4* ls = (float4*)l;
  for (int i = threadIdx.x; i < 1024; i += 256) ls[i] = gs[i];   // 16384 / 16
}
#endif

// ---------------- WMMA GEMM core: LDS-staged A tile vs 4 B tiles ----------------
// Ag: A fragments for (b,mt), contiguous KT*512 halfs. sA: 2*CHUNK_KT*512 halfs LDS.
template <int KT>
__device__ __forceinline__ void gemm_acc(const _Float16* __restrict__ Ag,
                                         const _Float16* __restrict__ Bw,
                                         int nt0, _Float16* sA, v8f acc[4]) {
  constexpr int NCH = KT / CHUNK_KT;     // 1 (K=512) or 4 (K=2048)
  const int lane = threadIdx.x & 31;
#if HAVE_TDM
  if (threadIdx.x < 32) tdm_load_a16k(Ag, sA);   // wave0 only (execz-skip for others)
#else
  coop_copy16k(Ag, sA);
#endif
  for (int c = 0; c < NCH; ++c) {
    const _Float16* Ab = sA + (size_t)(c & 1) * (CHUNK_KT * 512);
#if HAVE_TDM
    if (c + 1 < NCH) {
      if (threadIdx.x < 32)
        tdm_load_a16k(Ag + (size_t)(c + 1) * (CHUNK_KT * 512),
                      sA + (size_t)((c + 1) & 1) * (CHUNK_KT * 512));
#if HAVE_WAIT_TENSOR
      __builtin_amdgcn_s_wait_tensorcnt(1);      // oldest (chunk c) retired
#endif
    } else {
#if HAVE_WAIT_TENSOR
      __builtin_amdgcn_s_wait_tensorcnt(0);
#endif
    }
    __syncthreads();                             // chunk c visible to all waves
#else
    if (c > 0) coop_copy16k(Ag + (size_t)c * (CHUNK_KT * 512), (_Float16*)Ab);
    __syncthreads();
#endif
    for (int kt = 0; kt < CHUNK_KT; ++kt) {
      v16h av = ((const v16h*)(Ab + ((size_t)kt << 9)))[lane];
#pragma unroll
      for (int t = 0; t < 4; ++t) {
        const _Float16* Bt = Bw + (((size_t)(nt0 + t) * KT + (c * CHUNK_KT + kt)) << 9);
        v16h bv = ((const v16h*)Bt)[lane];
        acc[t] = __builtin_amdgcn_wmma_f32_16x16x32_f16(false, av, false, bv,
                                                        (short)0, acc[t], false, false);
      }
    }
    __syncthreads();                             // done reading buf before TDM reuses it
  }
}

// ---------------- weight prep: fp32 row-major [K][N] -> f16 B-fragments ----------------
__global__ void prep_weight_kernel(const float* __restrict__ W, _Float16* __restrict__ Wf,
                                   int K, int N) {
  int idx = blockIdx.x * 256 + threadIdx.x;
  if (idx >= K * N) return;
  int k = idx / N, n = idx % N;
  int KT = K >> 5;
  size_t o = (((size_t)(n >> 4) * KT + (k >> 5)) << 9) + bfrag_idx(k & 31, n & 15);
  Wf[o] = (_Float16)W[idx];
}

// ---------------- embed (tokens + pos enc) + LN1 ----------------
__global__ void embed_ln1_kernel(const float* __restrict__ x,
                                 const float* __restrict__ ctx,
                                 const float* __restrict__ regs,
                                 const float* __restrict__ g1,
                                 const float* __restrict__ b1n,
                                 float* __restrict__ xw,
                                 _Float16* __restrict__ actA) {
  int s = blockIdx.x, b = blockIdx.y, tid = threadIdx.x;
  __shared__ float red[256];
  __shared__ float row[DMODEL];
  size_t ro = ((size_t)b * SPAD + s) * DMODEL;
  int mt = s >> 4, r = s & 15;
  if (s >= S_TOT) {   // zero padded rows
    for (int j = 0; j < 2; ++j) {
      int d = tid * 2 + j;
      xw[ro + d] = 0.0f;
      actA[afrag_base(b, mt, d >> 5, KT1) + afrag_idx(r, d & 31)] = (_Float16)0.0f;
    }
    return;
  }
  for (int j = 0; j < 2; ++j) {
    int d = tid * 2 + j;
    float val;
    if (s < G_TOK) {
      float tok = (s == 0) ? ctx[d] : regs[(s - 1) * DMODEL + d];
      int i = d >> 1;
      float ang = (float)s * powf(10000.0f, -(2.0f * (float)i) / (float)DMODEL);
      val = tok + ((d & 1) ? cosf(ang) : sinf(ang));
    } else {
      val = x[((size_t)b * DMODEL + d) * 4096 + (size_t)(s - G_TOK)];
    }
    row[d] = val;
  }
  __syncthreads();
  float mean = block_reduce_sum256(row[tid * 2] + row[tid * 2 + 1], red) * (1.0f / DMODEL);
  float d0 = row[tid * 2] - mean, d1 = row[tid * 2 + 1] - mean;
  float var = block_reduce_sum256(d0 * d0 + d1 * d1, red) * (1.0f / DMODEL);
  float rstd = rsqrtf(var + 1e-5f);
  for (int j = 0; j < 2; ++j) {
    int d = tid * 2 + j;
    float nv = (row[d] - mean) * rstd * g1[d] + b1n[d];
    xw[ro + d] = nv;
    actA[afrag_base(b, mt, d >> 5, KT1) + afrag_idx(r, d & 31)] = (_Float16)nv;
  }
}

// ---------------- residual add + LN2 ----------------
__global__ void add_ln2_kernel(const float* __restrict__ xw,
                               float* __restrict__ attn,
                               const float* __restrict__ g2,
                               const float* __restrict__ b2n,
                               _Float16* __restrict__ hf) {
  int s = blockIdx.x, b = blockIdx.y, tid = threadIdx.x;
  __shared__ float red[256];
  __shared__ float row[DMODEL];
  size_t ro = ((size_t)b * SPAD + s) * DMODEL;
  int mt = s >> 4, r = s & 15;
  if (s >= S_TOT) {
    for (int j = 0; j < 2; ++j) {
      int d = tid * 2 + j;
      attn[ro + d] = 0.0f;
      hf[afrag_base(b, mt, d >> 5, KT1) + afrag_idx(r, d & 31)] = (_Float16)0.0f;
    }
    return;
  }
  for (int j = 0; j < 2; ++j) {
    int d = tid * 2 + j;
    float val = attn[ro + d] + xw[ro + d];
    attn[ro + d] = val;       // attn := attn + xw  (reference residual)
    row[d] = val;
  }
  __syncthreads();
  float mean = block_reduce_sum256(row[tid * 2] + row[tid * 2 + 1], red) * (1.0f / DMODEL);
  float d0 = row[tid * 2] - mean, d1 = row[tid * 2 + 1] - mean;
  float var = block_reduce_sum256(d0 * d0 + d1 * d1, red) * (1.0f / DMODEL);
  float rstd = rsqrtf(var + 1e-5f);
  for (int j = 0; j < 2; ++j) {
    int d = tid * 2 + j;
    float nv = (row[d] - mean) * rstd * g2[d] + b2n[d];
    hf[afrag_base(b, mt, d >> 5, KT1) + afrag_idx(r, d & 31)] = (_Float16)nv;
  }
}

// ---------------- GEMM: rows out fp32 (Q/K/V projections), N=512 ----------------
__global__ void gemm_rowout_kernel(const _Float16* __restrict__ A,
                                   const _Float16* __restrict__ Bw,
                                   const float* __restrict__ bias,
                                   float* __restrict__ C) {
  __shared__ _Float16 sA[2 * CHUNK_KT * 512];
  int w = blockIdx.x * 8 + (threadIdx.x >> 5);
  int lane = threadIdx.x & 31;
  int ng = w & 7;                       // 8 groups of 4 N-tiles (N=512); same mt per block
  int mt = (w >> 3) % MTILES;
  int b  = w / (8 * MTILES);
  v8f acc[4] = {{0,0,0,0,0,0,0,0},{0,0,0,0,0,0,0,0},{0,0,0,0,0,0,0,0},{0,0,0,0,0,0,0,0}};
  gemm_acc<KT1>(A + afrag_base(b, mt, 0, KT1), Bw, ng * 4, sA, acc);
  int lanehi = lane >> 4, lanelo = lane & 15;
#pragma unroll
  for (int t = 0; t < 4; ++t) {
    int col = (ng * 4 + t) * 16 + lanelo;
    float bc = bias[col];
#pragma unroll
    for (int j = 0; j < 8; ++j) {
      int rrow = mt * 16 + (lanehi << 3) + j;
      C[((size_t)b * SPAD + rrow) * DMODEL + col] = acc[t][j] + bc;
    }
  }
}

// ---------------- GEMM: MLP1 (N=2048) + SiLU -> f16 A-frags for MLP2 ----------------
__global__ void gemm_mlp1_kernel(const _Float16* __restrict__ A,
                                 const _Float16* __restrict__ Bw,
                                 const float* __restrict__ bias,
                                 _Float16* __restrict__ h1f) {
  __shared__ _Float16 sA[2 * CHUNK_KT * 512];
  int w = blockIdx.x * 8 + (threadIdx.x >> 5);
  int lane = threadIdx.x & 31;
  int ng = w & 31;                      // 32 groups of 4 N-tiles (N=2048); same mt per block
  int mt = (w >> 5) % MTILES;
  int b  = w / (32 * MTILES);
  v8f acc[4] = {{0,0,0,0,0,0,0,0},{0,0,0,0,0,0,0,0},{0,0,0,0,0,0,0,0},{0,0,0,0,0,0,0,0}};
  gemm_acc<KT1>(A + afrag_base(b, mt, 0, KT1), Bw, ng * 4, sA, acc);
  int lanehi = lane >> 4, lanelo = lane & 15;
  size_t hb = ((size_t)b * MTILES + mt) * KT2 << 9;
#pragma unroll
  for (int t = 0; t < 4; ++t) {
    int col = (ng * 4 + t) * 16 + lanelo;
    float bc = bias[col];
#pragma unroll
    for (int j = 0; j < 8; ++j) {
      int r = (lanehi << 3) + j;
      float vv = acc[t][j] + bc;
      float sv = vv / (1.0f + __expf(-vv));           // SiLU
      h1f[hb + ((size_t)(col >> 5) << 9) + afrag_idx(r, col & 31)] = (_Float16)sv;
    }
  }
}

// ---------------- GEMM: MLP2 (K=2048, N=512) + bias + residual + output scatter ------
__global__ void gemm_mlp2_kernel(const _Float16* __restrict__ A,
                                 const _Float16* __restrict__ Bw,
                                 const float* __restrict__ bias,
                                 const float* __restrict__ attn,
                                 float* __restrict__ out) {
  __shared__ _Float16 sA[2 * CHUNK_KT * 512];
  int w = blockIdx.x * 8 + (threadIdx.x >> 5);
  int lane = threadIdx.x & 31;
  int ng = w & 7;                       // 8 groups of 4 N-tiles (N=512); same mt per block
  int mt = (w >> 3) % MTILES;
  int b  = w / (8 * MTILES);
  v8f acc[4] = {{0,0,0,0,0,0,0,0},{0,0,0,0,0,0,0,0},{0,0,0,0,0,0,0,0},{0,0,0,0,0,0,0,0}};
  gemm_acc<KT2>(A + ((((size_t)b * MTILES + mt) * KT2) << 9), Bw, ng * 4, sA, acc);
  int lanehi = lane >> 4, lanelo = lane & 15;
#pragma unroll
  for (int t = 0; t < 4; ++t) {
    int col = (ng * 4 + t) * 16 + lanelo;
    float bc = bias[col];
#pragma unroll
    for (int j = 0; j < 8; ++j) {
      int rrow = mt * 16 + (lanehi << 3) + j;
      if (rrow >= S_TOT) continue;
      size_t ro = ((size_t)b * SPAD + rrow) * DMODEL;
      float val = acc[t][j] + bc + attn[ro + col];
      if (rrow >= G_TOK) {
        out[((size_t)b * DMODEL + col) * 4096 + (size_t)(rrow - G_TOK)] = val;
      } else if (rrow == 0) {
        out[FM_SIZE + (size_t)b * DMODEL + col] = val;
      } else {
        out[FM_SIZE + (size_t)BATCH * DMODEL +
            ((size_t)b * (G_TOK - 1) + (rrow - 1)) * DMODEL + col] = val;
      }
    }
  }
}

// ---------------- attention: G global queries, full softmax over S ----------------
__global__ void attn_global_kernel(const float* __restrict__ q,
                                   const float* __restrict__ k,
                                   const float* __restrict__ v,
                                   float* __restrict__ attn) {
  int blk = blockIdx.x;                  // b*H*G + h*G + g
  int g = blk % G_TOK;
  int h = (blk / G_TOK) % HEADS;
  int b = blk / (G_TOK * HEADS);
  int tid = threadIdx.x;
  __shared__ float qs[64];
  __shared__ float sc[S_TOT + 3];
  __shared__ float red[256];
  size_t hb = (size_t)b * SPAD * DMODEL + (size_t)h * 64;
  if (tid < 64) qs[tid] = q[hb + (size_t)g * DMODEL + tid];
  __syncthreads();
  for (int s = tid; s < S_TOT; s += 256) {
    const float* kr = k + hb + (size_t)s * DMODEL;
    float dot = 0.0f;
#pragma unroll
    for (int d = 0; d < 64; ++d) dot += qs[d] * kr[d];
    sc[s] = dot * 0.125f;
  }
  __syncthreads();
  float m = -1e30f;
  for (int s = tid; s < S_TOT; s += 256) m = fmaxf(m, sc[s]);
  red[tid] = m; __syncthreads();
  for (int off = 128; off > 0; off >>= 1) {
    if (tid < off) red[tid] = fmaxf(red[tid], red[tid + off]);
    __syncthreads();
  }
  m = red[0]; __syncthreads();
  float ssum = 0.0f;
  for (int s = tid; s < S_TOT; s += 256) { float e = __expf(sc[s] - m); sc[s] = e; ssum += e; }
  float denom = block_reduce_sum256(ssum, red);
  int d = tid & 63, quarter = tid >> 6;
  float acc = 0.0f;
  for (int s = quarter; s < S_TOT; s += 4) acc += sc[s] * v[hb + (size_t)s * DMODEL + d];
  red[tid] = acc; __syncthreads();
  if (quarter == 0) {
    float tot = red[d] + red[64 + d] + red[128 + d] + red[192 + d];
    attn[hb + (size_t)g * DMODEL + d] = tot / denom;
  }
}

// ---------------- attention: feature tokens (G globals + self), linear time ----------
__global__ void attn_feat_kernel(const float* __restrict__ q,
                                 const float* __restrict__ k,
                                 const float* __restrict__ v,
                                 float* __restrict__ attn) {
  int idx = blockIdx.x * 256 + threadIdx.x;   // b*H*4096
  int n = idx & 4095;
  int h = (idx >> 12) & (HEADS - 1);
  int b = idx >> 15;
  size_t hb = (size_t)b * SPAD * DMODEL + (size_t)h * 64;
  size_t rowo = hb + (size_t)(G_TOK + n) * DMODEL;
  float qv[64];
#pragma unroll
  for (int d = 0; d < 64; ++d) qv[d] = q[rowo + d];
  float sc[G_TOK + 1];
#pragma unroll
  for (int g = 0; g < G_TOK; ++g) {
    const float* kr = k + hb + (size_t)g * DMODEL;
    float dot = 0.0f;
#pragma unroll
    for (int d = 0; d < 64; ++d) dot += qv[d] * kr[d];
    sc[g] = dot * 0.125f;
  }
  {
    const float* kr = k + rowo;
    float dot = 0.0f;
#pragma unroll
    for (int d = 0; d < 64; ++d) dot += qv[d] * kr[d];
    sc[G_TOK] = dot * 0.125f;
  }
  float m = sc[0];
#pragma unroll
  for (int i = 1; i <= G_TOK; ++i) m = fmaxf(m, sc[i]);
  float den = 0.0f;
#pragma unroll
  for (int i = 0; i <= G_TOK; ++i) { sc[i] = __expf(sc[i] - m); den += sc[i]; }
  float inv = 1.0f / den;
  for (int d = 0; d < 64; ++d) {
    float o = sc[G_TOK] * v[rowo + d];
#pragma unroll
    for (int g = 0; g < G_TOK; ++g) o += sc[g] * v[hb + (size_t)g * DMODEL + d];
    attn[rowo + d] = o * inv;
  }
}

// =====================================================================================
extern "C" void kernel_launch(void* const* d_in, const int* in_sizes, int n_in,
                              void* d_out, int out_size, void* d_ws, size_t ws_size,
                              hipStream_t stream) {
  (void)in_sizes; (void)n_in; (void)out_size; (void)ws_size;
  const float* x    = (const float*)d_in[0];
  const float* ctx  = (const float*)d_in[1];
  const float* regs = (const float*)d_in[2];
  const float* Wq   = (const float*)d_in[3];
  const float* bq   = (const float*)d_in[4];
  const float* Wk   = (const float*)d_in[5];
  const float* bk   = (const float*)d_in[6];
  const float* Wv   = (const float*)d_in[7];
  const float* bv   = (const float*)d_in[8];
  const float* ln1g = (const float*)d_in[9];
  const float* ln1b = (const float*)d_in[10];
  const float* ln2g = (const float*)d_in[11];
  const float* ln2b = (const float*)d_in[12];
  const float* W1   = (const float*)d_in[13];
  const float* b1   = (const float*)d_in[14];
  const float* W2   = (const float*)d_in[15];
  const float* b2   = (const float*)d_in[16];
  float* out = (float*)d_out;

  // workspace carve-up (256B aligned)
  char* ws = (char*)d_ws;
  size_t off = 0;
  auto carve = [&](size_t bytes) -> void* {
    off = (off + 255) & ~(size_t)255;
    void* p = ws + off;
    off += bytes;
    return p;
  };
  const size_t NBS = (size_t)BATCH * SPAD * DMODEL;       // 16,842,752
  _Float16* wq_f = (_Float16*)carve((size_t)DMODEL * DMODEL * 2);
  _Float16* wk_f = (_Float16*)carve((size_t)DMODEL * DMODEL * 2);
  _Float16* wv_f = (_Float16*)carve((size_t)DMODEL * DMODEL * 2);
  _Float16* w1_f = (_Float16*)carve((size_t)DMODEL * DFF * 2);
  _Float16* w2_f = (_Float16*)carve((size_t)DFF * DMODEL * 2);
  float*    xw   = (float*)   carve(NBS * 4);
  _Float16* actA = (_Float16*)carve(NBS * 2);
  float*    qb   = (float*)   carve(NBS * 4);
  float*    kb   = (float*)   carve(NBS * 4);
  float*    vb   = (float*)   carve(NBS * 4);
  float*    attn = (float*)   carve(NBS * 4);
  _Float16* hf   = (_Float16*)carve(NBS * 2);
  _Float16* h1f  = (_Float16*)carve((size_t)BATCH * SPAD * DFF * 2);

  // 1) weight prep into WMMA B-fragment layout
  prep_weight_kernel<<<(DMODEL * DMODEL + 255) / 256, 256, 0, stream>>>(Wq, wq_f, DMODEL, DMODEL);
  prep_weight_kernel<<<(DMODEL * DMODEL + 255) / 256, 256, 0, stream>>>(Wk, wk_f, DMODEL, DMODEL);
  prep_weight_kernel<<<(DMODEL * DMODEL + 255) / 256, 256, 0, stream>>>(Wv, wv_f, DMODEL, DMODEL);
  prep_weight_kernel<<<(DMODEL * DFF + 255) / 256, 256, 0, stream>>>(W1, w1_f, DMODEL, DFF);
  prep_weight_kernel<<<(DFF * DMODEL + 255) / 256, 256, 0, stream>>>(W2, w2_f, DFF, DMODEL);

  // 2) embed + LN1 -> xw (fp32) + actA (f16 A-frags)
  embed_ln1_kernel<<<dim3(SPAD, BATCH), 256, 0, stream>>>(x, ctx, regs, ln1g, ln1b, xw, actA);

  // 3) Q/K/V WMMA GEMMs: waves = B*257*8 = 16448 -> 2056 blocks of 8 waves
  gemm_rowout_kernel<<<(BATCH * MTILES * 8) / 8, 256, 0, stream>>>(actA, wq_f, bq, qb);
  gemm_rowout_kernel<<<(BATCH * MTILES * 8) / 8, 256, 0, stream>>>(actA, wk_f, bk, kb);
  gemm_rowout_kernel<<<(BATCH * MTILES * 8) / 8, 256, 0, stream>>>(actA, wv_f, bv, vb);

  // 4) attention
  attn_global_kernel<<<BATCH * HEADS * G_TOK, 256, 0, stream>>>(qb, kb, vb, attn);
  attn_feat_kernel<<<(BATCH * HEADS * 4096) / 256, 256, 0, stream>>>(qb, kb, vb, attn);

  // 5) residual + LN2 -> attn (fp32, now attn+xw) + hf (f16 A-frags)
  add_ln2_kernel<<<dim3(SPAD, BATCH), 256, 0, stream>>>(xw, attn, ln2g, ln2b, hf);

  // 6) MLP1 (N=2048) + SiLU -> h1f frags: waves = B*257*32 -> 8224 blocks
  gemm_mlp1_kernel<<<(BATCH * MTILES * 32) / 8, 256, 0, stream>>>(hf, w1_f, b1, h1f);

  // 7) MLP2 (K=2048) + bias + residual, scatter to outputs: 2056 blocks
  gemm_mlp2_kernel<<<(BATCH * MTILES * 8) / 8, 256, 0, stream>>>(h1f, w2_f, b2, attn, out);
}
